// TransLayer_65395172049440
// MI455X (gfx1250) — matrix operand order repaired and checked by reading, not compiled, not run on hardware
//
#include <hip/hip_runtime.h>
#include <hip/hip_bf16.h>
#include <math.h>

typedef __bf16 bf16_t;
typedef __attribute__((ext_vector_type(16))) __bf16 v16bf;
typedef __attribute__((ext_vector_type(8)))  float  v8f;

// ---- problem constants ----
#define NB    2
#define NL    16384
#define NC    512
#define NHEAD 8
#define NHD   64
#define NBW   128     // windows
#define NTOK  256     // tokens per window
#define NMLP  2048
#define NROWS 32768   // B*L

// ---- workspace layout (bf16 element offsets) ----
static const size_t XN_OFF  = 0;                          // [32768,512]; reused as attention output o
static const size_t MN_OFF  = 16777216;                   // [128,8,256,64]; reused as hn
static const size_t SN_OFF  = 33554432;
static const size_t Q_OFF   = 50331648;                   // [128,8,256,64], pre-scaled 0.0625
static const size_t K_OFF   = 67108864;                   // [128,8,256,64]
static const size_t VT_OFF  = 83886080;                   // [128,8,64,256] (transposed)
static const size_t PA_OFF  = 100663296;                  // [128,8,256,64], pre-scaled 0.0625
static const size_t PB_OFF  = 117440512;
static const size_t WQ_OFF  = 134217728;                  // qkv_wT  [1536,512]
static const size_t WP_OFF  = WQ_OFF  + (size_t)1536*512; // proj_wT [512,512]
static const size_t WF1_OFF = WP_OFF  + (size_t)512*512;  // fc1_wT  [2048,512]
static const size_t WF2_OFF = WF1_OFF + (size_t)2048*512; // fc2_wT  [512,2048]

// ---------------- WMMA helpers (layouts per CDNA5 ISA 7.12.2) ----------------
__device__ __forceinline__ v8f vzero8() {
  v8f z;
  #pragma unroll
  for (int i = 0; i < 8; ++i) z[i] = 0.f;
  return z;
}

__device__ __forceinline__ v8f wmma_bf16(v16bf a, v16bf b, v8f c) {
  return __builtin_amdgcn_wmma_f32_16x16x32_bf16(false, a, false, b, (short)0, c, false, false);
}

// A fragment 16x32 bf16, row-major source with leading dim ld.
// lane m = lane&15; lo lanes: K=k0+0..7 & k0+16..23 ; hi lanes: K=k0+8..15 & k0+24..31
__device__ __forceinline__ v16bf frag_a(const bf16_t* base, int row0, int ld, int k0) {
  int lane = threadIdx.x & 31;
  int m = lane & 15;
  int kb = k0 + ((lane >> 4) << 3);
  const bf16_t* p = base + (size_t)(row0 + m) * ld + kb;
  union { v16bf v; uint4 q[2]; } u;
  u.q[0] = *(const uint4*)(p);
  u.q[1] = *(const uint4*)(p + 16);
  return u.v;
}

// B fragment 32x16: element B(kk,n) read from transposed storage T[(col0+n)*ldT + kk].
// lane n = lane&15; lo lanes: K=k0+0..7 & k0+16..23 ; hi lanes: K=k0+8..15 & k0+24..31
__device__ __forceinline__ v16bf frag_bT(const bf16_t* baseT, int col0, int ldT, int k0) {
  int lane = threadIdx.x & 31;
  int n = lane & 15;
  int kb = k0 + ((lane >> 4) << 3);
  const bf16_t* p = baseT + (size_t)(col0 + n) * ldT + kb;
  union { v16bf v; uint4 q[2]; } u;
  u.q[0] = *(const uint4*)(p);
  u.q[1] = *(const uint4*)(p + 16);
  return u.v;
}

__device__ __forceinline__ float wave_sum(float v) {
  #pragma unroll
  for (int o = 16; o > 0; o >>= 1) v += __shfl_xor(v, o, 32);
  return v;
}
__device__ __forceinline__ float wave_max(float v) {
  #pragma unroll
  for (int o = 16; o > 0; o >>= 1) v = fmaxf(v, __shfl_xor(v, o, 32));
  return v;
}

// ---------------- weight transpose + bf16 convert ----------------
__global__ void transpose_to_bf16(const float* __restrict__ src, bf16_t* __restrict__ dst,
                                  int rows, int cols) {
  int i = blockIdx.x * 256 + threadIdx.x;
  if (i >= rows * cols) return;
  int r = i / cols, c = i - r * cols;
  dst[(size_t)c * rows + r] = (bf16_t)src[i];
}

// ---------------- LN(x,m,s) + window partition ----------------
__global__ void __launch_bounds__(256)
ln_partition_kernel(const float* __restrict__ x, const float* __restrict__ m,
                    const float* __restrict__ s,
                    const float* g1, const float* b1, const float* g2, const float* b2,
                    const float* g3, const float* b3,
                    bf16_t* __restrict__ xn, bf16_t* __restrict__ mn, bf16_t* __restrict__ sn) {
  __shared__ float red[20];
  int t = blockIdx.x;                 // global token
  int b = t >> 14;
  int l = t & 16383;
  int row = l >> 7, col = l & 127;
  int hr = row >> 4, rr = row & 15, wr = col >> 4, cc = col & 15;
  int w = b * 64 + hr * 8 + wr;
  int n = rr * 16 + cc;
  int tid = threadIdx.x, wave = tid >> 5, lane = tid & 31;

  const float* bases[3] = { x + (size_t)t * NC, m + (size_t)t * NC, s + (size_t)t * NC };
  const float* gg[3] = { g1, g2, g3 };
  const float* bb[3] = { b1, b2, b3 };

  for (int e = 0; e < 3; ++e) {
    float v0 = bases[e][tid], v1 = bases[e][tid + 256];
    float sm = wave_sum(v0 + v1);
    float sq = wave_sum(v0 * v0 + v1 * v1);
    if (lane == 0) { red[wave] = sm; red[8 + wave] = sq; }
    __syncthreads();
    if (tid == 0) {
      float a = 0.f, q = 0.f;
      for (int i = 0; i < 8; ++i) { a += red[i]; q += red[8 + i]; }
      red[16] = a; red[17] = q;
    }
    __syncthreads();
    float mu = red[16] * (1.f / 512.f);
    float var = red[17] * (1.f / 512.f) - mu * mu;
    float rstd = rsqrtf(var + 1e-5f);
    #pragma unroll
    for (int p = 0; p < 2; ++p) {
      int ch = tid + p * 256;
      float v = p ? v1 : v0;
      bf16_t o = (bf16_t)((v - mu) * rstd * gg[e][ch] + bb[e][ch]);
      if (e == 0) {
        xn[((size_t)w * NTOK + n) * NC + ch] = o;
      } else {
        int h = ch >> 6, hd = ch & 63;
        size_t idx = (((size_t)(w * 8 + h)) * NTOK + n) * NHD + hd;
        if (e == 1) mn[idx] = o; else sn[idx] = o;
      }
    }
    __syncthreads();
  }
}

// ---------------- LN4 of residual stream ----------------
__global__ void __launch_bounds__(256)
ln4_kernel(const float* __restrict__ xin, const float* g, const float* bb,
           bf16_t* __restrict__ hn) {
  __shared__ float red[20];
  int t = blockIdx.x;
  int tid = threadIdx.x, wave = tid >> 5, lane = tid & 31;
  const float* p = xin + (size_t)t * NC;
  float v0 = p[tid], v1 = p[tid + 256];
  float sm = wave_sum(v0 + v1);
  float sq = wave_sum(v0 * v0 + v1 * v1);
  if (lane == 0) { red[wave] = sm; red[8 + wave] = sq; }
  __syncthreads();
  if (tid == 0) {
    float a = 0.f, q = 0.f;
    for (int i = 0; i < 8; ++i) { a += red[i]; q += red[8 + i]; }
    red[16] = a; red[17] = q;
  }
  __syncthreads();
  float mu = red[16] * (1.f / 512.f);
  float var = red[17] * (1.f / 512.f) - mu * mu;
  float rstd = rsqrtf(var + 1e-5f);
  hn[(size_t)t * NC + tid]       = (bf16_t)((v0 - mu) * rstd * g[tid] + bb[tid]);
  hn[(size_t)t * NC + tid + 256] = (bf16_t)((v1 - mu) * rstd * g[tid + 256] + bb[tid + 256]);
}

// ---------------- depthwise K=3 conv over tokens (per head) ----------------
__global__ void __launch_bounds__(256)
dwconv_kernel(const bf16_t* __restrict__ mn, const bf16_t* __restrict__ sn,
              const float* w1, const float* b1, const float* w2, const float* b2,
              bf16_t* __restrict__ pa, bf16_t* __restrict__ pb) {
  int tid = threadIdx.x;
  int g = blockIdx.x * 4 + (tid >> 6);     // (w*8+h)*256 + n
  int hd = tid & 63;
  int n = g & 255;
  int h = (g >> 8) & 7;
  size_t base = (size_t)(g - n) * NHD;
  float a1 = 0.f, a2 = 0.f;
  #pragma unroll
  for (int dk = 0; dk < 3; ++dk) {
    int nn = n + dk - 1;
    if (nn < 0 || nn > 255) continue;
    size_t idx = base + (size_t)nn * NHD + hd;
    a1 += w1[h * 3 + dk] * (float)mn[idx];
    a2 += w2[h * 3 + dk] * (float)sn[idx];
  }
  size_t oidx = (size_t)g * NHD + hd;
  pa[oidx] = (bf16_t)((a1 + b1[h]) * 0.0625f);   // fold 0.5 * HD^-0.5
  pb[oidx] = (bf16_t)(a2 + b2[h]);
}

// ---------------- QKV GEMM: [32768,512] x [512,1536], scatter epilogue ----------------
__global__ void __launch_bounds__(256)
qkv_gemm_kernel(const bf16_t* __restrict__ xn, const bf16_t* __restrict__ wT,
                const float* __restrict__ bias,
                bf16_t* __restrict__ q, bf16_t* __restrict__ k, bf16_t* __restrict__ vT) {
  int wid = blockIdx.x * 8 + (threadIdx.x >> 5);
  int rt = wid / 24, cq = wid - rt * 24;
  int lane = threadIdx.x & 31;
  int cn = lane & 15, rbase = (lane >> 4) << 3;
  v8f acc[4];
  #pragma unroll
  for (int c = 0; c < 4; ++c) acc[c] = vzero8();
  for (int k0 = 0; k0 < NC; k0 += 32) {
    v16bf a = frag_a(xn, rt * 16, NC, k0);
    if (k0 + 32 < NC)
      __builtin_prefetch(xn + (size_t)(rt * 16 + cn) * NC + k0 + 32, 0, 0);
    #pragma unroll
    for (int c = 0; c < 4; ++c)
      acc[c] = wmma_bf16(a, frag_bT(wT, cq * 64 + c * 16, NC, k0), acc[c]);
  }
  #pragma unroll
  for (int c = 0; c < 4; ++c) {
    int colbase = cq * 64 + c * 16;
    int three = colbase >> 9;
    int head  = (colbase >> 6) & 7;
    #pragma unroll
    for (int r = 0; r < 8; ++r) {
      int gr = rt * 16 + rbase + r;
      int w = gr >> 8, n = gr & 255;
      int col = colbase + cn;
      int hd = col & 63;
      float val = acc[c][r] + bias[col];
      if (three == 0)      q[(((size_t)(w * 8 + head)) * NTOK + n) * NHD + hd] = (bf16_t)(val * 0.0625f);
      else if (three == 1) k[(((size_t)(w * 8 + head)) * NTOK + n) * NHD + hd] = (bf16_t)val;
      else                 vT[(((size_t)(w * 8 + head)) * NHD + hd) * NTOK + n] = (bf16_t)val;
    }
  }
}

// ---------------- attention: one block per (window, head) ----------------
__global__ void __launch_bounds__(256)
attn_kernel(const bf16_t* __restrict__ q, const bf16_t* __restrict__ k,
            const bf16_t* __restrict__ vT, const bf16_t* __restrict__ pa,
            const bf16_t* __restrict__ pb, bf16_t* __restrict__ o) {
  extern __shared__ char smem[];
  float*  Sf = (float*)smem;                              // 256*256 f32 = 256KB
  bf16_t* Pc = (bf16_t*)(smem + 65536 * sizeof(float));   // 256*64 bf16 = 32KB
  int blk = blockIdx.x;
  int w = blk >> 3, h = blk & 7;
  size_t base = ((size_t)(w * 8 + h)) * NTOK * NHD;       // also valid for vT ([64,256])
  const bf16_t* qb  = q  + base;
  const bf16_t* kb  = k  + base;
  const bf16_t* pab = pa + base;
  const bf16_t* pbb = pb + base;
  const bf16_t* vtb = vT + base;
  int tid = threadIdx.x, wave = tid >> 5, lane = tid & 31;
  int cn = lane & 15, rbase = (lane >> 4) << 3;

  // S = q'·k^T + pa'·pb^T   (q', pa' pre-scaled by 0.5*HD^-0.5)
  for (int t = wave; t < 256; t += 8) {
    int tr = t >> 4, tc = t & 15;
    v8f acc = vzero8();
    #pragma unroll
    for (int ks = 0; ks < NHD; ks += 32)
      acc = wmma_bf16(frag_a(qb, tr * 16, NHD, ks), frag_bT(kb, tc * 16, NHD, ks), acc);
    #pragma unroll
    for (int ks = 0; ks < NHD; ks += 32)
      acc = wmma_bf16(frag_a(pab, tr * 16, NHD, ks), frag_bT(pbb, tc * 16, NHD, ks), acc);
    #pragma unroll
    for (int r = 0; r < 8; ++r)
      Sf[(tr * 16 + rbase + r) * 256 + tc * 16 + cn] = acc[r];
  }
  __syncthreads();

  // softmax rows (wave32 shuffle reductions), 32 rows per wave
  for (int rr = 0; rr < 32; ++rr) {
    int row = wave * 32 + rr;
    float vals[8];
    float mx = -3.4e38f;
    #pragma unroll
    for (int j = 0; j < 8; ++j) { vals[j] = Sf[row * 256 + j * 32 + lane]; mx = fmaxf(mx, vals[j]); }
    mx = wave_max(mx);
    float sum = 0.f;
    #pragma unroll
    for (int j = 0; j < 8; ++j) { vals[j] = __expf(vals[j] - mx); sum += vals[j]; }
    sum = wave_sum(sum);
    float inv = 1.f / sum;
    #pragma unroll
    for (int j = 0; j < 8; ++j) Sf[row * 256 + j * 32 + lane] = vals[j] * inv;
  }
  __syncthreads();

  // O = P · v, chunked over token-cols of P (64 at a time through LDS bf16)
  v8f oacc[8];
  #pragma unroll
  for (int i = 0; i < 8; ++i) oacc[i] = vzero8();
  for (int nc = 0; nc < 4; ++nc) {
    for (int i = tid; i < 256 * 64; i += 256) {
      int ri = i >> 6, ci = i & 63;
      Pc[ri * 64 + ci] = (bf16_t)Sf[ri * 256 + nc * 64 + ci];
    }
    __syncthreads();
    #pragma unroll
    for (int rsl = 0; rsl < 2; ++rsl) {
      #pragma unroll
      for (int ks = 0; ks < 64; ks += 32) {
        v16bf a = frag_a(Pc, (wave * 2 + rsl) * 16, 64, ks);
        #pragma unroll
        for (int cs = 0; cs < 4; ++cs)
          oacc[rsl * 4 + cs] = wmma_bf16(a, frag_bT(vtb, cs * 16, NTOK, nc * 64 + ks),
                                         oacc[rsl * 4 + cs]);
      }
    }
    __syncthreads();
  }
  #pragma unroll
  for (int rsl = 0; rsl < 2; ++rsl)
    #pragma unroll
    for (int cs = 0; cs < 4; ++cs)
      #pragma unroll
      for (int r = 0; r < 8; ++r) {
        int n = wave * 32 + rsl * 16 + rbase + r;
        int hd = cs * 16 + cn;
        o[((size_t)(w * NTOK + n)) * NC + h * NHD + hd] = (bf16_t)oacc[rsl * 4 + cs][r];
      }
}

// ---------------- proj GEMM + un-partition + x residual ----------------
__global__ void __launch_bounds__(256)
proj_gemm_kernel(const bf16_t* __restrict__ o, const bf16_t* __restrict__ wT,
                 const float* __restrict__ bias, const float* __restrict__ x,
                 float* __restrict__ out) {
  int wid = blockIdx.x * 8 + (threadIdx.x >> 5);
  int rt = wid >> 3, cq = wid & 7;
  int lane = threadIdx.x & 31;
  int cn = lane & 15, rbase = (lane >> 4) << 3;
  v8f acc[4];
  #pragma unroll
  for (int c = 0; c < 4; ++c) acc[c] = vzero8();
  for (int k0 = 0; k0 < NC; k0 += 32) {
    v16bf a = frag_a(o, rt * 16, NC, k0);
    #pragma unroll
    for (int c = 0; c < 4; ++c)
      acc[c] = wmma_bf16(a, frag_bT(wT, cq * 64 + c * 16, NC, k0), acc[c]);
  }
  #pragma unroll
  for (int c = 0; c < 4; ++c) {
    #pragma unroll
    for (int r = 0; r < 8; ++r) {
      int gr = rt * 16 + rbase + r;          // window-major row
      int w = gr >> 8, n = gr & 255;
      int b = w >> 6, hr = (w >> 3) & 7, wr = w & 7;
      int rr = n >> 4, c2 = n & 15;
      int l = (hr * 16 + rr) * 128 + wr * 16 + c2;
      int col = cq * 64 + c * 16 + cn;
      size_t addr = ((size_t)b * NL + l) * NC + col;
      out[addr] = x[addr] + acc[c][r] + bias[col];
    }
  }
}

// ---------------- fused MLP: fc1 + exact GELU + fc2 + residual ----------------
__global__ void __launch_bounds__(256)
mlp_kernel(const bf16_t* __restrict__ hn, const bf16_t* __restrict__ w1T,
           const float* __restrict__ b1, const bf16_t* __restrict__ w2T,
           const float* __restrict__ b2, float* __restrict__ out) {
  __shared__ __align__(16) bf16_t hsm[32 * 64];
  int row0 = blockIdx.x * 32;
  int tid = threadIdx.x, wave = tid >> 5, lane = tid & 31;
  int cn = lane & 15, rbase = (lane >> 4) << 3;
  int rs1 = wave >> 2, cs1 = wave & 3;   // fc1 tile map (2x4)
  int rs2 = wave & 1,  cg  = wave >> 1;  // fc2 map: row strip + 128-col group
  v8f acc2[8];
  #pragma unroll
  for (int i = 0; i < 8; ++i) acc2[i] = vzero8();

  for (int cc = 0; cc < NMLP / 64; ++cc) {
    // fc1: 32x64 chunk of h, one 16x16 tile per wave
    v8f a1 = vzero8();
    for (int k0 = 0; k0 < NC; k0 += 32)
      a1 = wmma_bf16(frag_a(hn, row0 + rs1 * 16, NC, k0),
                     frag_bT(w1T, cc * 64 + cs1 * 16, NC, k0), a1);
    #pragma unroll
    for (int r = 0; r < 8; ++r) {
      int lrow = rs1 * 16 + rbase + r;
      int lcol = cs1 * 16 + cn;
      int jg = cc * 64 + lcol;
      float v = a1[r] + b1[jg];
      float gel = 0.5f * v * (1.f + erff(v * 0.70710678118654752f));
      hsm[lrow * 64 + lcol] = (bf16_t)gel;
    }
    __syncthreads();
    // fc2 accumulate: h chunk (32x64) x w2 chunk (64x512)
    #pragma unroll
    for (int ks = 0; ks < 64; ks += 32) {
      v16bf a = frag_a(hsm, rs2 * 16, 64, ks);
      #pragma unroll
      for (int t = 0; t < 8; ++t)
        acc2[t] = wmma_bf16(a, frag_bT(w2T, cg * 128 + t * 16, NMLP, cc * 64 + ks), acc2[t]);
    }
    __syncthreads();
  }
  #pragma unroll
  for (int t = 0; t < 8; ++t) {
    int col0 = cg * 128 + t * 16;
    #pragma unroll
    for (int r = 0; r < 8; ++r) {
      int R = row0 + rs2 * 16 + rbase + r;
      int col = col0 + cn;
      size_t addr = (size_t)R * NC + col;
      out[addr] = out[addr] + acc2[t][r] + b2[col];
    }
  }
}

// ---------------- host launcher ----------------
extern "C" void kernel_launch(void* const* d_in, const int* in_sizes, int n_in,
                              void* d_out, int out_size, void* d_ws, size_t ws_size,
                              hipStream_t stream) {
  const float* x      = (const float*)d_in[0];
  const float* m      = (const float*)d_in[1];
  const float* s      = (const float*)d_in[2];
  const float* ln1_g  = (const float*)d_in[3];
  const float* ln1_b  = (const float*)d_in[4];
  const float* ln2_g  = (const float*)d_in[5];
  const float* ln2_b  = (const float*)d_in[6];
  const float* ln3_g  = (const float*)d_in[7];
  const float* ln3_b  = (const float*)d_in[8];
  const float* ln4_g  = (const float*)d_in[9];
  const float* ln4_b  = (const float*)d_in[10];
  const float* qkv_w  = (const float*)d_in[11];
  const float* qkv_b  = (const float*)d_in[12];
  const float* proj_w = (const float*)d_in[13];
  const float* proj_b = (const float*)d_in[14];
  const float* pe1_w  = (const float*)d_in[15];
  const float* pe1_b  = (const float*)d_in[16];
  const float* pe2_w  = (const float*)d_in[17];
  const float* pe2_b  = (const float*)d_in[18];
  const float* fc1_w  = (const float*)d_in[19];
  const float* fc1_b  = (const float*)d_in[20];
  const float* fc2_w  = (const float*)d_in[21];
  const float* fc2_b  = (const float*)d_in[22];
  float* out = (float*)d_out;
  bf16_t* ws = (bf16_t*)d_ws;

  // 1) weights -> bf16 transposed
  transpose_to_bf16<<<(512 * 1536 + 255) / 256, 256, 0, stream>>>(qkv_w, ws + WQ_OFF, 512, 1536);
  transpose_to_bf16<<<(512 * 512 + 255) / 256, 256, 0, stream>>>(proj_w, ws + WP_OFF, 512, 512);
  transpose_to_bf16<<<(512 * 2048 + 255) / 256, 256, 0, stream>>>(fc1_w, ws + WF1_OFF, 512, 2048);
  transpose_to_bf16<<<(2048 * 512 + 255) / 256, 256, 0, stream>>>(fc2_w, ws + WF2_OFF, 2048, 512);
  // 2) LN + window partition
  ln_partition_kernel<<<NROWS, 256, 0, stream>>>(x, m, s, ln1_g, ln1_b, ln2_g, ln2_b,
                                                 ln3_g, ln3_b,
                                                 ws + XN_OFF, ws + MN_OFF, ws + SN_OFF);
  // 3) depthwise conv for positional term
  dwconv_kernel<<<NBW * NHEAD * NTOK / 4, 256, 0, stream>>>(ws + MN_OFF, ws + SN_OFF,
                                                            pe1_w, pe1_b, pe2_w, pe2_b,
                                                            ws + PA_OFF, ws + PB_OFF);
  // 4) QKV GEMM
  qkv_gemm_kernel<<<(2048 * 24) / 8, 256, 0, stream>>>(ws + XN_OFF, ws + WQ_OFF, qkv_b,
                                                       ws + Q_OFF, ws + K_OFF, ws + VT_OFF);
  // 5) attention (288KB dynamic LDS; CDNA5 WGP allows up to 320KB)
  attn_kernel<<<NBW * NHEAD, 256, 65536 * sizeof(float) + 256 * 64 * sizeof(bf16_t), stream>>>(
      ws + Q_OFF, ws + K_OFF, ws + VT_OFF, ws + PA_OFF, ws + PB_OFF, ws + XN_OFF /* o aliases xn */);
  // 6) proj + residual into d_out
  proj_gemm_kernel<<<(2048 * 8) / 8, 256, 0, stream>>>(ws + XN_OFF, ws + WP_OFF, proj_b, x, out);
  // 7) LN4 -> hn (aliases mn)
  ln4_kernel<<<NROWS, 256, 0, stream>>>(out, ln4_g, ln4_b, ws + MN_OFF);
  // 8) fused MLP + residual
  mlp_kernel<<<NROWS / 32, 256, 0, stream>>>(ws + MN_OFF, ws + WF1_OFF, fc1_b,
                                             ws + WF2_OFF, fc2_b, out);
}
